// BuildK_25005299597348
// MI455X (gfx1250) — matrix operand order repaired and checked by reading, not compiled, not run on hardware
//
#include <hip/hip_runtime.h>
#include <hip/hip_bf16.h>
#include <stdint.h>

typedef __attribute__((ext_vector_type(16))) _Float16 v16h;
typedef __attribute__((ext_vector_type(8)))  _Float16 v8h;
typedef __attribute__((ext_vector_type(8)))  float    v8f;

// ds_swizzle, group-of-32 mode: offset = (xor<<10)|(or<<5)|and, bit15=0
#define SWZ(v, off) __int_as_float(__builtin_amdgcn_ds_swizzle(__float_as_int(v), (off)))

__device__ __forceinline__ float wave_max32(float v) {
  v = fmaxf(v, SWZ(v, 0x041F));  // xor 1
  v = fmaxf(v, SWZ(v, 0x081F));  // xor 2
  v = fmaxf(v, SWZ(v, 0x101F));  // xor 4
  v = fmaxf(v, SWZ(v, 0x201F));  // xor 8
  v = fmaxf(v, SWZ(v, 0x401F));  // xor 16
  return v;
}
__device__ __forceinline__ float wave_sum32(float v) {
  v += SWZ(v, 0x041F);
  v += SWZ(v, 0x081F);
  v += SWZ(v, 0x101F);
  v += SWZ(v, 0x201F);
  v += SWZ(v, 0x401F);
  return v;
}

// 8-to-1 select of c[lane&7]; all source lanes in a half hold identical c,
// so a cndmask tree on lane-id bits is exact and spill-free.
__device__ __forceinline__ float sel8(v8f c, int lane) {
  bool b0 = (lane & 1) != 0, b1 = (lane & 2) != 0, b2 = (lane & 4) != 0;
  float s01 = b0 ? c[1] : c[0];
  float s23 = b0 ? c[3] : c[2];
  float s45 = b0 ? c[5] : c[4];
  float s67 = b0 ? c[7] : c[6];
  float a03 = b1 ? s23 : s01;
  float a47 = b1 ? s67 : s45;
  return b2 ? a47 : a03;
}

// ---------------------------------------------------------------------------
// Pass 1: transpose input1 [32, N] -> uut [N, 32] (f16, 64B rows), and
// precompute squared norms from the f16-rounded values (consistent with WMMA).
// ---------------------------------------------------------------------------
__global__ void __launch_bounds__(256)
prep_transpose(const float* __restrict__ in1, _Float16* __restrict__ uut,
               float* __restrict__ norms, int N) {
  __shared__ float tile[32][33];
  int x = threadIdx.x;           // 0..31 (pixel within tile on read, feature on write)
  int y = threadIdx.y;           // 0..7
  int p0 = blockIdx.x * 32;
#pragma unroll
  for (int i = 0; i < 4; ++i) {
    int f = y + 8 * i;
    float v = (p0 + x < N) ? in1[(size_t)f * N + p0 + x] : 0.f;
    _Float16 h = (_Float16)v;          // round once; reuse rounded value everywhere
    tile[f][x] = (float)h;
  }
  __syncthreads();
#pragma unroll
  for (int i = 0; i < 4; ++i) {
    int p = y + 8 * i;
    if (p0 + p < N) uut[(size_t)(p0 + p) * 32 + x] = (_Float16)tile[x][p];  // coalesced 2B stores
  }
  if (y == 0 && p0 + x < N) {
    float s = 0.f;
#pragma unroll
    for (int f = 0; f < 32; ++f) { float v = tile[f][x]; s = fmaf(v, v, s); }
    norms[p0 + x] = s;
  }
}

// ---------------------------------------------------------------------------
// Pass 2: one wave per pixel j. 3x v_wmma_f32_16x16x32_f16 compute all 48
// dot(u_j, u_nbr); epilogue distributes 48 slots over 32 lanes, applies
// msd -> -sqrt -> softmax with ds_swizzle butterflies, coalesced stores.
// ---------------------------------------------------------------------------
__global__ void __launch_bounds__(256)
knn_softmax(const _Float16* __restrict__ uut, const float* __restrict__ norms,
            const long long* __restrict__ idx, float* __restrict__ out, int J) {
  int lane = threadIdx.x & 31;
  int wave = threadIdx.x >> 5;
  int j = blockIdx.x * 8 + wave;
  if (j >= J) return;                  // whole wave exits together; EXEC stays full
  int half = lane >> 4;                // which K-half this lane supplies
  int m15  = lane & 15;

  // B-matrix: u_j broadcast into all 16 columns. Lanes 0-15 hold K=0..15,
  // lanes 16-31 hold K=16..31 (contiguous K order, same as A). 32B load, broadcast-friendly.
  v16h bv = *(const v16h*)(uut + (size_t)j * 32 + half * 16);

  const size_t ibase = (size_t)j * 48;
  long long R0, R1, R2;
  v8f C0, C1, C2;
#pragma unroll
  for (int t = 0; t < 3; ++t) {
    // A-matrix tile t: rows = neighbors 16t..16t+15. Lane l supplies row
    // (l&15); K = 0..15 for lanes<16, 16..31 for lanes>=16.
    long long r = idx[ibase + 16 * t + m15];
    const _Float16* rp = uut + (size_t)r * 32 + half * 16;
    v8h lo = *(const v8h*)(rp);        // K = off..off+7      (b128)
    v8h hi = *(const v8h*)(rp + 8);    // K = off+8..off+15   (b128)
    v16h a;
#pragma unroll
    for (int i = 0; i < 8; ++i) { a[i] = lo[i]; a[8 + i] = hi[i]; }
    v8f cz = {};
    v8f c = __builtin_amdgcn_wmma_f32_16x16x32_f16(
        false, a, false, bv, (short)0, cz, false, false);
    if (t == 0) { R0 = r; C0 = c; }
    else if (t == 1) { R1 = r; C1 = c; }
    else { R2 = r; C2 = c; }
  }

  // --- De-replicate: lane l owns slot1 = l, slot2 = l+32 (lanes<16 only) ---
  // Slot s=16t+m': dot lives in C_t[m'&7] within half (m'>>3), replicated
  // across that half's 16 lanes.
  v8f ct;
#pragma unroll
  for (int i = 0; i < 8; ++i) ct[i] = half ? C1[i] : C0[i];  // t = lane>=16 ? 1 : 0
  float d1 = sel8(ct, lane);
  float d2 = sel8(C2, lane);
  bool swap = (((lane >> 3) ^ (lane >> 4)) & 1) != 0;  // needed half != own half
  float d1x = SWZ(d1, 0x401F);
  float d2x = SWZ(d2, 0x401F);
  d1 = swap ? d1x : d1;
  d2 = swap ? d2x : d2;

  // A-phase index registers already hold exactly the rows for our slots
  // (replicated in both halves) -> zero index reloads.
  int r1 = (int)(half ? R1 : R0);
  int r2 = (int)R2;
  float nj  = norms[j];
  float nk1 = norms[r1];
  float nk2 = norms[r2];

  const float kInv32 = 1.0f / 32.0f, kEps = 1e-9f, kLog2e = 1.44269504088896341f;
  float msd1 = fmaxf(fmaf(-2.0f, d1, nj + nk1), 0.0f) * kInv32 + kEps;
  float msd2 = fmaxf(fmaf(-2.0f, d2, nj + nk2), 0.0f) * kInv32 + kEps;
  float D1 = -sqrtf(msd1);
  float D2 = -sqrtf(msd2);
  if (lane >= 16) D2 = -__builtin_inff();  // slot2 invalid for lanes>=16

  float mx = wave_max32(fmaxf(D1, D2));
  float e1 = exp2f((D1 - mx) * kLog2e);
  float e2 = exp2f((D2 - mx) * kLog2e);   // exp2(-inf) = 0 for invalid slots
  float S  = wave_sum32(e1 + e2);
  float inv = __builtin_amdgcn_rcpf(S);
  inv = inv * (2.0f - S * inv);           // one Newton step

  float* orow = out + (size_t)j * 48;
  orow[lane] = e1 * inv;                  // k = 0..31, fully coalesced
  if (lane < 16) orow[32 + lane] = e2 * inv;  // k = 32..47
}

// ---------------------------------------------------------------------------
extern "C" void kernel_launch(void* const* d_in, const int* in_sizes, int n_in,
                              void* d_out, int out_size, void* d_ws, size_t ws_size,
                              hipStream_t stream) {
  const float*     in1 = (const float*)d_in[0];       // [1,32,384,384] f32
  const long long* idx = (const long long*)d_in[1];   // [N,48] int64
  float* out = (float*)d_out;                         // [N,48] f32

  int N = in_sizes[0] / 32;       // pixels (rows of UU)
  int J = in_sizes[1] / 48;       // query rows (== N here)

  // Workspace: [norms: N f32][uut: N*32 f16]  (~10 MB total)
  float*    norms = (float*)d_ws;
  _Float16* uut   = (_Float16*)((char*)d_ws + (size_t)N * sizeof(float));

  prep_transpose<<<(N + 31) / 32, dim3(32, 8), 0, stream>>>(in1, uut, norms, N);
  knn_softmax<<<(J + 7) / 8, 256, 0, stream>>>(uut, norms, idx, out, J);
}
